// SelfAttention_20933670601473
// MI455X (gfx1250) — matrix operand order repaired and checked by reading, not compiled, not run on hardware
//
#include <hip/hip_runtime.h>
#include <hip/hip_bf16.h>

// ---------------------------------------------------------------------------
// Causal multi-head self-attention forward for MI455X (gfx1250, wave32).
// Pipeline: pack(f32->bf16, transpose weights) -> QKV GEMM (WMMA bf16)
//           -> flash attention (WMMA bf16, online softmax, async V staging)
//           -> output projection GEMM (WMMA bf16, f32 out).
// ---------------------------------------------------------------------------

typedef __bf16 bf16_t;
typedef __attribute__((ext_vector_type(16))) __bf16 v16bf;
typedef __attribute__((ext_vector_type(8)))  __bf16 v8bf;
typedef __attribute__((ext_vector_type(8)))  float  v8f;

#define B_SZ   4
#define T_SEQ  2048
#define C_DIM  1024
#define H_NUM  16
#define HD     64
#define M_ROWS (B_SZ * T_SEQ)          // 8192
#define N_QKV  (3 * C_DIM)             // 3072

#define XB_N   ((long)M_ROWS * C_DIM)            // 8388608
#define WQ_N   ((long)C_DIM * N_QKV)             // 3145728
#define WO_N   ((long)C_DIM * C_DIM)             // 1048576
#define QKV_N  ((long)B_SZ * H_NUM * T_SEQ * HD) // 8388608 per tensor

static __device__ __forceinline__ v8f wmma_bf16(v16bf a, v16bf b, v8f c) {
  return __builtin_amdgcn_wmma_f32_16x16x32_bf16(false, a, false, b,
                                                 (short)0, c, false, false);
}

// Build a 16x32 bf16 A-fragment from a row-major matrix row.
// Lane m=lane&15 holds row m; kbase=(lane>>4)*16;
// elements 0..15 = K[kbase..kbase+15] (contiguous, matches B fragments).
static __device__ __forceinline__ v16bf load_a_frag(const bf16_t* row, int kba) {
  v8bf lo = *(const v8bf*)(row + 2 * kba);
  v8bf hi = *(const v8bf*)(row + 2 * kba + 8);
  v16bf a;
#pragma unroll
  for (int i = 0; i < 8; ++i) { a[i] = lo[i]; a[8 + i] = hi[i]; }
  return a;
}

// 16-byte async copy: global memory -> LDS, tracked by ASYNCcnt.
static __device__ __forceinline__ void async_copy_b128(uint32_t lds_off,
                                                       const void* gptr) {
  asm volatile("global_load_async_to_lds_b128 %0, %1, off"
               :: "v"(lds_off), "v"(gptr) : "memory");
}

static __device__ __forceinline__ void wait_async0() {
  asm volatile("s_wait_asynccnt 0x0" ::: "memory");
}

// ---------------------------------------------------------------------------
// Kernel 1: pack/convert.  xb = bf16(x);  wqkvT[n][k] = w_qkv[k][n];
//           woutT[n][k] = w_out[k][n]   (all bf16)
// ---------------------------------------------------------------------------
__global__ __launch_bounds__(256) void pack_kernel(
    const float* __restrict__ x, const float* __restrict__ wqkv,
    const float* __restrict__ wout, bf16_t* __restrict__ xb,
    bf16_t* __restrict__ wqkvT, bf16_t* __restrict__ woutT) {
  long i = (long)blockIdx.x * 256 + threadIdx.x;
  const long total = XB_N + WQ_N + WO_N;
  if (i >= total) return;
  if (i < XB_N) {
    xb[i] = (bf16_t)x[i];
  } else if (i < XB_N + WQ_N) {
    long j = i - XB_N;
    int n = (int)(j >> 10), k = (int)(j & 1023);
    wqkvT[j] = (bf16_t)wqkv[(long)k * N_QKV + n];
  } else {
    long j = i - XB_N - WQ_N;
    int n = (int)(j >> 10), k = (int)(j & 1023);
    woutT[j] = (bf16_t)wout[(long)k * C_DIM + n];
  }
}

// ---------------------------------------------------------------------------
// Kernel 2: QKV projection GEMM.  qkv[m][n] = sum_k xb[m][k]*wqkvT[n][k]+b[n]
// Each wave computes a 16x64 tile (A-fragment reused across 4 N-tiles).
// Scatter epilogue writes Q (scaled by 1/sqrt(D)) and K in [B,H,T,D],
// V transposed in [B,H,D,T] (so P@V B-fragments are contiguous).
// ---------------------------------------------------------------------------
__global__ __launch_bounds__(128) void qkv_gemm(
    const bf16_t* __restrict__ xb, const bf16_t* __restrict__ wT,
    const float* __restrict__ bias, bf16_t* __restrict__ q,
    bf16_t* __restrict__ kk, bf16_t* __restrict__ vT) {
  const int lane = threadIdx.x & 31;
  const int wave = threadIdx.x >> 5;
  const int m0 = blockIdx.y * 64 + wave * 16;
  const int n0 = blockIdx.x * 64;
  const int mrow = lane & 15;
  const int kba = (lane >> 4) * 8;
  const int kbb = (lane >> 4) * 16;

  v8f acc[4] = {v8f{}, v8f{}, v8f{}, v8f{}};
  const bf16_t* arow = xb + (long)(m0 + mrow) * C_DIM;
  for (int k = 0; k < C_DIM; k += 32) {
    v16bf a = load_a_frag(arow + k, kba);
#pragma unroll
    for (int t = 0; t < 4; ++t) {
      v16bf b = *(const v16bf*)(wT + (long)(n0 + t * 16 + mrow) * C_DIM + k + kbb);
      acc[t] = wmma_bf16(a, b, acc[t]);
    }
  }

  const int rowb = m0 + (lane < 16 ? 0 : 8);
#pragma unroll
  for (int t = 0; t < 4; ++t) {
    const int n = n0 + t * 16 + (lane & 15);
    const float bv = bias[n];
    const int which = n >> 10;      // 0=Q 1=K 2=V
    const int c = n & 1023;
    const int h = c >> 6, d = c & 63;
#pragma unroll
    for (int r = 0; r < 8; ++r) {
      const int m = rowb + r;
      const int bb = m >> 11, tt = m & 2047;
      const long bh = (long)bb * H_NUM + h;
      const float v = acc[t][r] + bv;
      if (which == 0)      q [(bh * T_SEQ + tt) * HD + d] = (bf16_t)(v * 0.125f);
      else if (which == 1) kk[(bh * T_SEQ + tt) * HD + d] = (bf16_t)v;
      else                 vT[(bh * HD + d) * T_SEQ + tt] = (bf16_t)v;
    }
  }
}

// ---------------------------------------------------------------------------
// Kernel 3: causal flash attention.  One wave = 16 query rows, whole D=64.
// V tiles are prefetched into LDS one 32-key step ahead with
// global_load_async_to_lds_b128 (ASYNCcnt), double-buffered per wave.
// ---------------------------------------------------------------------------
#define SV_TILE (64 * 32)   // one V stage buffer: 64 d-rows x 32 keys (bf16)

// Stage one 64x32 V^T tile (4 KB) into LDS asynchronously.
// vT rows are d (stride T_SEQ); each row contributes 64B = 4 x 16B chunks.
static __device__ __forceinline__ void stage_v_async(
    bf16_t* dst, const bf16_t* vbase, int kb, int lane) {
#pragma unroll
  for (int i = 0; i < 8; ++i) {
    const int lin = i * 32 + lane;        // 0..255 16-byte chunks
    const int d   = lin >> 2;             // 0..63
    const int ce  = (lin & 3) * 8;        // column offset in elements
    const uint32_t lds_off = (uint32_t)(uintptr_t)(dst + d * 32 + ce);
    async_copy_b128(lds_off, vbase + (long)d * T_SEQ + kb + ce);
  }
}

__global__ __launch_bounds__(128) void attn_kernel(
    const bf16_t* __restrict__ q, const bf16_t* __restrict__ kmat,
    const bf16_t* __restrict__ vT, bf16_t* __restrict__ ob) {
  __shared__ alignas(32) bf16_t svbuf[4][2][SV_TILE];  // 32 KB async V stage
  __shared__ alignas(32) bf16_t pbuf[4][16 * 32];      //  4 KB P staging

  const int lane = threadIdx.x & 31;
  const int wave = threadIdx.x >> 5;
  const int bh = blockIdx.y;                        // 0..B*H-1
  const int bb = bh >> 4, h = bh & 15;
  const int q0 = blockIdx.x * 64 + wave * 16;
  const int mrow = lane & 15;
  const int kba = (lane >> 4) * 8;
  const int kbb = (lane >> 4) * 16;

  const bf16_t* qbase = q    + (long)bh * T_SEQ * HD;
  const bf16_t* kbase = kmat + (long)bh * T_SEQ * HD;
  const bf16_t* vbase = vT   + (long)bh * HD * T_SEQ;

  // Q fragments for the two 32-wide K-dim steps (D = 64). Kept resident.
  v16bf qf[2];
#pragma unroll
  for (int s = 0; s < 2; ++s)
    qf[s] = load_a_frag(qbase + (long)(q0 + mrow) * HD + s * 32, kba);

  v8f o[4] = {v8f{}, v8f{}, v8f{}, v8f{}};
  float mrun[8], lrun[8];
#pragma unroll
  for (int r = 0; r < 8; ++r) { mrun[r] = -INFINITY; lrun[r] = 0.0f; }

  const int kend = q0 + 16;                         // per-wave causal bound
  bf16_t* svw = &svbuf[wave][0][0];

  // Prefetch first V tile before entering the loop.
  stage_v_async(svw, vbase, 0, lane);

  int bufi = 0;
  for (int kb = 0; kb < kend; kb += 32, bufi ^= 1) {
    // ---- S = Q K^T for 16 rows x 32 keys (two 16-wide key tiles) ----
    v8f s0 = v8f{}, s1 = v8f{};
#pragma unroll
    for (int s = 0; s < 2; ++s) {
      v16bf k0 = *(const v16bf*)(kbase + (long)(kb      + mrow) * HD + s * 32 + kbb);
      v16bf k1 = *(const v16bf*)(kbase + (long)(kb + 16 + mrow) * HD + s * 32 + kbb);
      s0 = wmma_bf16(qf[s], k0, s0);
      s1 = wmma_bf16(qf[s], k1, s1);
    }

    // Current V tile is resident after this; issue next prefetch.
    wait_async0();
    if (kb + 32 < kend)
      stage_v_async(svw + (bufi ^ 1) * SV_TILE, vbase, kb + 32, lane);

    // ---- causal mask + online softmax in C-register layout ----
    const int col0 = kb + (lane & 15);
    const int col1 = col0 + 16;
    const int rhalf = (lane < 16 ? 0 : 8);
    float alpha[8];
#pragma unroll
    for (int r = 0; r < 8; ++r) {
      const int row = q0 + rhalf + r;
      float a0 = (col0 <= row) ? s0[r] : -INFINITY;
      float a1 = (col1 <= row) ? s1[r] : -INFINITY;
      float mx = fmaxf(a0, a1);
      mx = fmaxf(mx, __shfl_xor(mx, 1, 32));
      mx = fmaxf(mx, __shfl_xor(mx, 2, 32));
      mx = fmaxf(mx, __shfl_xor(mx, 4, 32));
      mx = fmaxf(mx, __shfl_xor(mx, 8, 32));
      const float mnew = fmaxf(mrun[r], mx);
      const float al = __expf(mrun[r] - mnew);
      const float p0 = __expf(a0 - mnew);
      const float p1 = __expf(a1 - mnew);
      float sum = p0 + p1;
      sum += __shfl_xor(sum, 1, 32);
      sum += __shfl_xor(sum, 2, 32);
      sum += __shfl_xor(sum, 4, 32);
      sum += __shfl_xor(sum, 8, 32);
      lrun[r] = lrun[r] * al + sum;
      mrun[r] = mnew;
      alpha[r] = al;
      s0[r] = p0;
      s1[r] = p1;
    }

    // ---- rescale O accumulators ----
#pragma unroll
    for (int t = 0; t < 4; ++t)
#pragma unroll
      for (int r = 0; r < 8; ++r) o[t][r] *= alpha[r];

    // ---- P (C-layout f32) -> LDS bf16 -> A-fragment layout ----
    bf16_t* pl = pbuf[wave];
    {
      const int colA = lane & 15;
#pragma unroll
      for (int r = 0; r < 8; ++r) {
        pl[(rhalf + r) * 32 + colA]      = (bf16_t)s0[r];
        pl[(rhalf + r) * 32 + 16 + colA] = (bf16_t)s1[r];
      }
    }
    asm volatile("s_wait_dscnt 0x0" ::: "memory");  // RAW on per-wave LDS
    v16bf pf = load_a_frag(pl + mrow * 32, kba);

    // ---- O += P @ V from the staged LDS tile ----
    const bf16_t* svt = svw + bufi * SV_TILE;
#pragma unroll
    for (int t = 0; t < 4; ++t) {
      v16bf vf = *(const v16bf*)(svt + (t * 16 + mrow) * 32 + kbb);
      o[t] = wmma_bf16(pf, vf, o[t]);
    }
  }

  // ---- normalize and store as bf16 [B,T,C] for the output GEMM ----
#pragma unroll
  for (int r = 0; r < 8; ++r) lrun[r] = 1.0f / lrun[r];
  const int rhalf = (lane < 16 ? 0 : 8);
#pragma unroll
  for (int t = 0; t < 4; ++t) {
    const int d = t * 16 + (lane & 15);
#pragma unroll
    for (int r = 0; r < 8; ++r) {
      const int tq = q0 + rhalf + r;
      ob[((long)bb * T_SEQ + tq) * C_DIM + h * HD + d] = (bf16_t)(o[t][r] * lrun[r]);
    }
  }
}

// ---------------------------------------------------------------------------
// Kernel 4: output projection.  out[m][n] = sum_k ob[m][k]*woutT[n][k]+b[n]
// ---------------------------------------------------------------------------
__global__ __launch_bounds__(128) void out_gemm(
    const bf16_t* __restrict__ ob, const bf16_t* __restrict__ wT,
    const float* __restrict__ bias, float* __restrict__ out) {
  const int lane = threadIdx.x & 31;
  const int wave = threadIdx.x >> 5;
  const int m0 = blockIdx.y * 64 + wave * 16;
  const int n0 = blockIdx.x * 64;
  const int mrow = lane & 15;
  const int kba = (lane >> 4) * 8;
  const int kbb = (lane >> 4) * 16;

  v8f acc[4] = {v8f{}, v8f{}, v8f{}, v8f{}};
  const bf16_t* arow = ob + (long)(m0 + mrow) * C_DIM;
  for (int k = 0; k < C_DIM; k += 32) {
    v16bf a = load_a_frag(arow + k, kba);
#pragma unroll
    for (int t = 0; t < 4; ++t) {
      v16bf b = *(const v16bf*)(wT + (long)(n0 + t * 16 + mrow) * C_DIM + k + kbb);
      acc[t] = wmma_bf16(a, b, acc[t]);
    }
  }

  const int rowb = m0 + (lane < 16 ? 0 : 8);
#pragma unroll
  for (int t = 0; t < 4; ++t) {
    const int n = n0 + t * 16 + (lane & 15);
    const float bv = bias[n];
#pragma unroll
    for (int r = 0; r < 8; ++r)
      out[(long)(rowb + r) * C_DIM + n] = acc[t][r] + bv;
  }
}

// ---------------------------------------------------------------------------
extern "C" void kernel_launch(void* const* d_in, const int* in_sizes, int n_in,
                              void* d_out, int out_size, void* d_ws, size_t ws_size,
                              hipStream_t stream) {
  (void)in_sizes; (void)n_in; (void)out_size; (void)ws_size;
  const float* x     = (const float*)d_in[0];
  const float* w_qkv = (const float*)d_in[1];
  const float* b_qkv = (const float*)d_in[2];
  const float* w_out = (const float*)d_in[3];
  const float* b_out = (const float*)d_in[4];
  float* out = (float*)d_out;

  char* ws = (char*)d_ws;
  size_t off = 0;
  auto alloc = [&](long nbytes) {
    void* p = ws + off;
    off = (off + (size_t)nbytes + 255) & ~(size_t)255;
    return p;
  };
  bf16_t* xb     = (bf16_t*)alloc(XB_N * 2);   // 16 MB
  bf16_t* wqkvT  = (bf16_t*)alloc(WQ_N * 2);   // 6 MB
  bf16_t* woutT  = (bf16_t*)alloc(WO_N * 2);   // 2 MB
  bf16_t* qd     = (bf16_t*)alloc(QKV_N * 2);  // 16 MB
  bf16_t* kd     = (bf16_t*)alloc(QKV_N * 2);  // 16 MB
  bf16_t* vTd    = (bf16_t*)alloc(QKV_N * 2);  // 16 MB
  bf16_t* obd    = (bf16_t*)alloc(XB_N * 2);   // 16 MB   (total ~88 MB)

  const long total = XB_N + WQ_N + WO_N;
  const int pack_blocks = (int)((total + 255) / 256);
  pack_kernel<<<pack_blocks, 256, 0, stream>>>(x, w_qkv, w_out, xb, wqkvT, woutT);

  qkv_gemm<<<dim3(N_QKV / 64, M_ROWS / 64), 128, 0, stream>>>(
      xb, wqkvT, b_qkv, qd, kd, vTd);

  attn_kernel<<<dim3(T_SEQ / 64, B_SZ * H_NUM), 128, 0, stream>>>(
      qd, kd, vTd, obd);

  out_gemm<<<dim3(C_DIM / 64, M_ROWS / 64), 128, 0, stream>>>(
      obd, woutT, b_out, out);
}
